// Attention_67405216743622
// MI455X (gfx1250) — matrix operand order repaired and checked by reading, not compiled, not run on hardware
//
#include <hip/hip_runtime.h>

typedef __attribute__((ext_vector_type(16))) _Float16 v16h;
typedef __attribute__((ext_vector_type(8)))  float    v8f;
typedef __attribute__((ext_vector_type(4)))  unsigned int u32x4;
typedef __attribute__((address_space(3)))    _Float16 lds_f16;

#define D_MODEL 768
#define NHEADS  12
#define D_HEAD  64
#define SEQ     2048
#define BATCH   4
#define TOKENS  (BATCH * SEQ)      // 8192
#define NQKV    (3 * D_MODEL)      // 2304
#define KSTEPS  (D_MODEL / 32)     // 24

__device__ __forceinline__ v8f wmma_f16(v16h a, v16h b, v8f c) {
  return __builtin_amdgcn_wmma_f32_16x16x32_f16(false, a, false, b, (short)0, c,
                                                false, false);
}

// A fragment: 16x32 f16 tile from row-major [rows x ld] at (row0, k0).
// Code's own contiguous k order (not the ISA table order): lane m=L&15, hi=L>>4; half e -> k = hi*16 + e (contiguous 32B).
__device__ __forceinline__ v16h load_A(const _Float16* base, int ld, int row0,
                                       int k0, int lane) {
  int m = lane & 15, hi = lane >> 4;
  const _Float16* p = base + (size_t)(row0 + m) * ld + k0 + hi * 16;
  union { v16h h; u32x4 q[2]; } f;
  f.q[0] = *(const u32x4*)(p);        // e=0..7 : k = hi*16 + e
  f.q[1] = *(const u32x4*)(p + 8);    // e=8..15: k = hi*16 + e
  return f.h;
}

// B fragment: 32x16 f16, B[k][n] = src[(col0+n)*ld + kbase + k].
// Code's own contiguous k order (not the ISA table order): lane n=L&15, hi=L>>4; half e -> k = hi*16 + e (contiguous 32B).
__device__ __forceinline__ v16h load_B_rows(const _Float16* base, int ld,
                                            int col0, int kbase, int lane) {
  int n = lane & 15, hi = lane >> 4;
  const _Float16* p = base + (size_t)(col0 + n) * ld + kbase + hi * 16;
  union { v16h h; u32x4 q[2]; } f;
  f.q[0] = *(const u32x4*)(p);
  f.q[1] = *(const u32x4*)(p + 8);
  return f.h;
}

// B fragment from pre-swizzled weights: [(kb*ntiles + nt)*32 + lane]*16 halves.
__device__ __forceinline__ v16h load_B_sw(const _Float16* swz, int ntiles,
                                          int kb, int nt, int lane) {
  const _Float16* p = swz + (((size_t)kb * ntiles + nt) * 32 + lane) * 16;
  union { v16h h; u32x4 q[2]; } f;
  f.q[0] = *(const u32x4*)(p);
  f.q[1] = *(const u32x4*)(p + 8);
  return f.h;
}

// ---- CDNA5 async global->LDS (ASYNCcnt) ----
__device__ __forceinline__ unsigned lds_off_of(const _Float16* p) {
  return (unsigned)(uintptr_t)(lds_f16*)p;
}
__device__ __forceinline__ void async_g2l_b128(unsigned lds_off,
                                               const _Float16* gptr) {
  asm volatile("global_load_async_to_lds_b128 %0, %1, off"
               :: "v"(lds_off), "v"((unsigned long long)(uintptr_t)gptr)
               : "memory");
}
__device__ __forceinline__ void wait_async_le8() {
  asm volatile("s_wait_asynccnt 0x8" ::: "memory");
}
__device__ __forceinline__ void wait_async_0() {
  asm volatile("s_wait_asynccnt 0x0" ::: "memory");
}

// ---------------- conversion / packing kernels ----------------

__global__ void k_cvt_x(const float* __restrict__ x, _Float16* __restrict__ xf,
                        int n) {
  int i = blockIdx.x * blockDim.x + threadIdx.x;
  if (i < n) xf[i] = (_Float16)x[i];
}

// Pack W_Q|W_K|W_V ([12,768,64] each) into B-swizzled f16 [768 x 2304].
__global__ void k_pack_wqkv(const float* __restrict__ WQ,
                            const float* __restrict__ WK,
                            const float* __restrict__ WV,
                            _Float16* __restrict__ sw) {
  int idx = blockIdx.x * blockDim.x + threadIdx.x;
  if (idx >= D_MODEL * NQKV) return;
  int e    = idx & 15;
  int lane = (idx >> 4) & 31;
  int tile = idx >> 9;                 // 512 halves per tile
  int nt   = tile % (NQKV / 16);
  int kb   = tile / (NQKV / 16);
  int k = kb * 32 + (lane >> 4) * 16 + e;   // d_model index
  int c = nt * 16 + (lane & 15);            // column 0..2303
  int p = c / D_MODEL, c2 = c % D_MODEL;
  int hh = c2 >> 6, dd = c2 & 63;
  const float* W = (p == 0) ? WQ : (p == 1) ? WK : WV;
  sw[idx] = (_Float16)W[((size_t)hh * D_MODEL + k) * D_HEAD + dd];
}

// Pack W_O ([12,64,768] == row-major [768 x 768]) into B-swizzled f16.
__global__ void k_pack_wo(const float* __restrict__ WO,
                          _Float16* __restrict__ sw) {
  int idx = blockIdx.x * blockDim.x + threadIdx.x;
  if (idx >= D_MODEL * D_MODEL) return;
  int e    = idx & 15;
  int lane = (idx >> 4) & 31;
  int tile = idx >> 9;
  int nt   = tile % (D_MODEL / 16);
  int kb   = tile / (D_MODEL / 16);
  int k = kb * 32 + (lane >> 4) * 16 + e;
  int c = nt * 16 + (lane & 15);
  sw[idx] = (_Float16)WO[(size_t)k * D_MODEL + c];
}

// ---------------- QKV projection GEMM ----------------
// [8192 x 768] * [768 x 2304]; block = 8 waves covering 32 rows x 256 cols.
// Explicit 2x-unrolled ping-pong: set0/set1 fragments, each refilled right
// after its WMMAs consume it (fresh defs -> no register copies).
__global__ __launch_bounds__(256) void k_qkv(
    const _Float16* __restrict__ xf, const _Float16* __restrict__ wsw,
    const float* __restrict__ bQ, const float* __restrict__ bK,
    const float* __restrict__ bV, _Float16* __restrict__ Qb,
    _Float16* __restrict__ Kb, _Float16* __restrict__ Vt) {
  const int NT = NQKV / 16;
  int lane = threadIdx.x & 31, w = threadIdx.x >> 5;
  int hi = lane >> 4, nn = lane & 15;
  int mw = w & 1, nw = w >> 1;
  int r0 = blockIdx.x * 32 + mw * 16;
  int ntbase = blockIdx.y * 16 + nw * 4;
  v8f acc[4] = {};
  v16h a0 = load_A(xf, D_MODEL, r0, 0, lane);
  v16h p00 = load_B_sw(wsw, NT, 0, ntbase + 0, lane);
  v16h p01 = load_B_sw(wsw, NT, 0, ntbase + 1, lane);
  v16h p02 = load_B_sw(wsw, NT, 0, ntbase + 2, lane);
  v16h p03 = load_B_sw(wsw, NT, 0, ntbase + 3, lane);
  v16h a1 = load_A(xf, D_MODEL, r0, 32, lane);
  v16h p10 = load_B_sw(wsw, NT, 1, ntbase + 0, lane);
  v16h p11 = load_B_sw(wsw, NT, 1, ntbase + 1, lane);
  v16h p12 = load_B_sw(wsw, NT, 1, ntbase + 2, lane);
  v16h p13 = load_B_sw(wsw, NT, 1, ntbase + 3, lane);
  for (int kb = 0; kb < KSTEPS; kb += 2) {
    int n2 = (kb + 2 < KSTEPS) ? kb + 2 : kb;
    int n3 = (kb + 3 < KSTEPS) ? kb + 3 : kb + 1;
    acc[0] = wmma_f16(a0, p00, acc[0]);
    acc[1] = wmma_f16(a0, p01, acc[1]);
    acc[2] = wmma_f16(a0, p02, acc[2]);
    acc[3] = wmma_f16(a0, p03, acc[3]);
    a0  = load_A(xf, D_MODEL, r0, n2 * 32, lane);
    p00 = load_B_sw(wsw, NT, n2, ntbase + 0, lane);
    p01 = load_B_sw(wsw, NT, n2, ntbase + 1, lane);
    p02 = load_B_sw(wsw, NT, n2, ntbase + 2, lane);
    p03 = load_B_sw(wsw, NT, n2, ntbase + 3, lane);
    acc[0] = wmma_f16(a1, p10, acc[0]);
    acc[1] = wmma_f16(a1, p11, acc[1]);
    acc[2] = wmma_f16(a1, p12, acc[2]);
    acc[3] = wmma_f16(a1, p13, acc[3]);
    a1  = load_A(xf, D_MODEL, r0, n3 * 32, lane);
    p10 = load_B_sw(wsw, NT, n3, ntbase + 0, lane);
    p11 = load_B_sw(wsw, NT, n3, ntbase + 1, lane);
    p12 = load_B_sw(wsw, NT, n3, ntbase + 2, lane);
    p13 = load_B_sw(wsw, NT, n3, ntbase + 3, lane);
  }
#pragma unroll
  for (int t = 0; t < 4; ++t) {
    int c = (ntbase + t) * 16 + nn;
    int p = c / D_MODEL, c2 = c % D_MODEL;
    int hh = c2 >> 6, dd = c2 & 63;
    float bias = (p == 0 ? bQ : p == 1 ? bK : bV)[hh * 64 + dd];
#pragma unroll
    for (int v = 0; v < 8; ++v) {
      int r = r0 + v + 8 * hi;
      int bb = r >> 11, ss = r & (SEQ - 1);
      float val = acc[t][v] + bias;
      size_t bh = (size_t)(bb * NHEADS + hh);
      if (p == 0)
        Qb[(bh * SEQ + ss) * 64 + dd] = (_Float16)(val * 0.125f);  // fold 1/sqrt(64)
      else if (p == 1)
        Kb[(bh * SEQ + ss) * 64 + dd] = (_Float16)val;
      else
        Vt[(bh * 64 + dd) * SEQ + ss] = (_Float16)val;             // transposed
    }
  }
}

// ---------------- flash attention ----------------
// block = 4 waves = one 64-row q tile of one (b,h); wave owns a 16-row strip.
// K/V chunks staged to LDS with async global->LDS (double buffered).
__global__ __launch_bounds__(128) void k_attn(
    const _Float16* __restrict__ Qb, const _Float16* __restrict__ Kb,
    const _Float16* __restrict__ Vt, _Float16* __restrict__ Zb) {
  __shared__ __align__(16) _Float16 Kst[2][64 * 64];  // [key][d]
  __shared__ __align__(16) _Float16 Vst[2][64 * 64];  // [d][key]
  __shared__ __align__(16) _Float16 Pst[4][16 * 64];
  int tid = threadIdx.x;
  int lane = tid & 31, w = tid >> 5;
  int hi = lane >> 4, nn = lane & 15;
  int bh = blockIdx.x >> 5, qt = blockIdx.x & 31;
  int bb = bh / NHEADS, h = bh % NHEADS;
  const _Float16* Qp = Qb + (size_t)bh * SEQ * 64;
  const _Float16* Kp = Kb + (size_t)bh * SEQ * 64;
  const _Float16* Vp = Vt + (size_t)bh * 64 * SEQ;
  int qb = qt * 64;
  int strip = qb + w * 16;

  v16h qa0 = load_A(Qp, 64, strip, 0, lane);
  v16h qa1 = load_A(Qp, 64, strip, 32, lane);
  v8f accO[4] = {};
  float rmax[8], rsum[8];
#pragma unroll
  for (int v = 0; v < 8; ++v) { rmax[v] = -3.0e38f; rsum[v] = 0.f; }
  _Float16* Pw = &Pst[w][0];

  // stage chunk 0 asynchronously: K chunk is 8KB contiguous; V chunk is 64
  // segments of 128B (transposed layout), both as per-lane b128 async ops.
  {
    unsigned ko = lds_off_of(&Kst[0][0]), vo = lds_off_of(&Vst[0][0]);
#pragma unroll
    for (int j = 0; j < 4; ++j) {
      int l = j * 128 + tid;
      async_g2l_b128(ko + l * 16, Kp + l * 8);
      async_g2l_b128(vo + l * 16, Vp + (size_t)(l >> 3) * SEQ + (l & 7) * 8);
    }
  }

  int nchunks = qb / 64 + 1;
  for (int i = 0; i < nchunks; ++i) {
    int k0 = i * 64;
    const _Float16* Kc = &Kst[i & 1][0];
    const _Float16* Vc = &Vst[i & 1][0];
    bool hasnext = (i + 1 < nchunks);
    if (hasnext) {
      int k1 = k0 + 64;
      unsigned ko = lds_off_of(&Kst[(i + 1) & 1][0]);
      unsigned vo = lds_off_of(&Vst[(i + 1) & 1][0]);
#pragma unroll
      for (int j = 0; j < 4; ++j) {
        int l = j * 128 + tid;
        async_g2l_b128(ko + l * 16, Kp + (size_t)k1 * 64 + l * 8);
        async_g2l_b128(vo + l * 16,
                       Vp + (size_t)(l >> 3) * SEQ + k1 + (l & 7) * 8);
      }
      wait_async_le8();   // oldest 8 (this chunk's ops) complete
    } else {
      wait_async_0();
    }
    __syncthreads();      // all waves' staging for chunk i visible

    v8f s[4] = {};
#pragma unroll
    for (int t = 0; t < 4; ++t) {
      s[t] = wmma_f16(qa0, load_B_rows(Kc, 64, t * 16, 0, lane), s[t]);
      s[t] = wmma_f16(qa1, load_B_rows(Kc, 64, t * 16, 32, lane), s[t]);
    }
    // causal mask (exactly -1e5 like reference) + online softmax per row
#pragma unroll
    for (int v = 0; v < 8; ++v) {
      int qg = strip + v + 8 * hi;
      float lm = -3.0e38f;
#pragma unroll
      for (int t = 0; t < 4; ++t) {
        int kk = k0 + t * 16 + nn;
        float x = s[t][v];
        if (kk > qg) x = -1.0e5f;
        s[t][v] = x;
        lm = fmaxf(lm, x);
      }
#pragma unroll
      for (int m = 1; m < 16; m <<= 1) lm = fmaxf(lm, __shfl_xor(lm, m, 32));
      float M = fmaxf(rmax[v], lm);
      float corr = __expf(rmax[v] - M);
      rmax[v] = M;
      float ls = 0.f;
#pragma unroll
      for (int t = 0; t < 4; ++t) {
        float pv = __expf(s[t][v] - M);
        s[t][v] = pv;
        ls += pv;
      }
#pragma unroll
      for (int m = 1; m < 16; m <<= 1) ls += __shfl_xor(ls, m, 32);
      rsum[v] = rsum[v] * corr + ls;
#pragma unroll
      for (int t = 0; t < 4; ++t) accO[t][v] = accO[t][v] * corr;
    }
    // stage P (C-layout -> row-major LDS) and reload as A fragments
#pragma unroll
    for (int t = 0; t < 4; ++t)
#pragma unroll
      for (int v = 0; v < 8; ++v)
        Pw[(v + 8 * hi) * 64 + t * 16 + nn] = (_Float16)s[t][v];
    v16h pa0 = load_A(Pw, 64, 0, 0, lane);
    v16h pa1 = load_A(Pw, 64, 0, 32, lane);
#pragma unroll
    for (int t = 0; t < 4; ++t) {
      accO[t] = wmma_f16(pa0, load_B_rows(Vc, 64, t * 16, 0, lane), accO[t]);
      accO[t] = wmma_f16(pa1, load_B_rows(Vc, 64, t * 16, 32, lane), accO[t]);
    }
    __syncthreads();      // all waves done reading buffers before re-stage
  }
  float inv[8];
#pragma unroll
  for (int v = 0; v < 8; ++v) inv[v] = 1.0f / rsum[v];
#pragma unroll
  for (int t = 0; t < 4; ++t)
#pragma unroll
    for (int v = 0; v < 8; ++v) {
      int qg = strip + v + 8 * hi;
      Zb[((size_t)bb * SEQ + qg) * D_MODEL + h * 64 + t * 16 + nn] =
          (_Float16)(accO[t][v] * inv[v]);
    }
}

// ---------------- output projection GEMM ----------------
__global__ __launch_bounds__(256) void k_oproj(
    const _Float16* __restrict__ Zb, const _Float16* __restrict__ wsw,
    const float* __restrict__ bO, float* __restrict__ out) {
  const int NT = D_MODEL / 16;
  int lane = threadIdx.x & 31, w = threadIdx.x >> 5;
  int hi = lane >> 4, nn = lane & 15;
  int mw = w & 1, nw = w >> 1;
  int r0 = blockIdx.x * 32 + mw * 16;
  int ntbase = blockIdx.y * 16 + nw * 4;
  v8f acc[4] = {};
  v16h a0 = load_A(Zb, D_MODEL, r0, 0, lane);
  v16h p00 = load_B_sw(wsw, NT, 0, ntbase + 0, lane);
  v16h p01 = load_B_sw(wsw, NT, 0, ntbase + 1, lane);
  v16h p02 = load_B_sw(wsw, NT, 0, ntbase + 2, lane);
  v16h p03 = load_B_sw(wsw, NT, 0, ntbase + 3, lane);
  v16h a1 = load_A(Zb, D_MODEL, r0, 32, lane);
  v16h p10 = load_B_sw(wsw, NT, 1, ntbase + 0, lane);
  v16h p11 = load_B_sw(wsw, NT, 1, ntbase + 1, lane);
  v16h p12 = load_B_sw(wsw, NT, 1, ntbase + 2, lane);
  v16h p13 = load_B_sw(wsw, NT, 1, ntbase + 3, lane);
  for (int kb = 0; kb < KSTEPS; kb += 2) {
    int n2 = (kb + 2 < KSTEPS) ? kb + 2 : kb;
    int n3 = (kb + 3 < KSTEPS) ? kb + 3 : kb + 1;
    acc[0] = wmma_f16(a0, p00, acc[0]);
    acc[1] = wmma_f16(a0, p01, acc[1]);
    acc[2] = wmma_f16(a0, p02, acc[2]);
    acc[3] = wmma_f16(a0, p03, acc[3]);
    a0  = load_A(Zb, D_MODEL, r0, n2 * 32, lane);
    p00 = load_B_sw(wsw, NT, n2, ntbase + 0, lane);
    p01 = load_B_sw(wsw, NT, n2, ntbase + 1, lane);
    p02 = load_B_sw(wsw, NT, n2, ntbase + 2, lane);
    p03 = load_B_sw(wsw, NT, n2, ntbase + 3, lane);
    acc[0] = wmma_f16(a1, p10, acc[0]);
    acc[1] = wmma_f16(a1, p11, acc[1]);
    acc[2] = wmma_f16(a1, p12, acc[2]);
    acc[3] = wmma_f16(a1, p13, acc[3]);
    a1  = load_A(Zb, D_MODEL, r0, n3 * 32, lane);
    p10 = load_B_sw(wsw, NT, n3, ntbase + 0, lane);
    p11 = load_B_sw(wsw, NT, n3, ntbase + 1, lane);
    p12 = load_B_sw(wsw, NT, n3, ntbase + 2, lane);
    p13 = load_B_sw(wsw, NT, n3, ntbase + 3, lane);
  }
#pragma unroll
  for (int t = 0; t < 4; ++t) {
    int c = (ntbase + t) * 16 + nn;
    float bias = bO[c];
#pragma unroll
    for (int v = 0; v < 8; ++v) {
      int r = r0 + v + 8 * hi;
      out[(size_t)r * D_MODEL + c] = acc[t][v] + bias;
    }
  }
}

// ---------------- launch ----------------

extern "C" void kernel_launch(void* const* d_in, const int* in_sizes, int n_in,
                              void* d_out, int out_size, void* d_ws,
                              size_t ws_size, hipStream_t stream) {
  const float* x  = (const float*)d_in[0];
  const float* WQ = (const float*)d_in[1];
  const float* WK = (const float*)d_in[2];
  const float* WV = (const float*)d_in[3];
  const float* WO = (const float*)d_in[4];
  const float* bQ = (const float*)d_in[5];
  const float* bK = (const float*)d_in[6];
  const float* bV = (const float*)d_in[7];
  const float* bO = (const float*)d_in[8];
  float* out = (float*)d_out;

  char* ws = (char*)d_ws;
  size_t off = 0;
  auto carve = [&](size_t bytes) {
    void* p = ws + off;
    off = (off + bytes + 255) & ~(size_t)255;
    return p;
  };
  _Float16* Xf   = (_Float16*)carve((size_t)TOKENS * D_MODEL * 2);
  _Float16* Wqkv = (_Float16*)carve((size_t)D_MODEL * NQKV * 2);
  _Float16* Wo   = (_Float16*)carve((size_t)D_MODEL * D_MODEL * 2);
  _Float16* Qb   = (_Float16*)carve((size_t)BATCH * NHEADS * SEQ * 64 * 2);
  _Float16* Kb   = (_Float16*)carve((size_t)BATCH * NHEADS * SEQ * 64 * 2);
  _Float16* Vt   = (_Float16*)carve((size_t)BATCH * NHEADS * SEQ * 64 * 2);
  _Float16* Zb   = (_Float16*)carve((size_t)TOKENS * D_MODEL * 2);

  int nx = TOKENS * D_MODEL;
  k_cvt_x<<<(nx + 255) / 256, 256, 0, stream>>>(x, Xf, nx);
  k_pack_wqkv<<<(D_MODEL * NQKV + 255) / 256, 256, 0, stream>>>(WQ, WK, WV, Wqkv);
  k_pack_wo<<<(D_MODEL * D_MODEL + 255) / 256, 256, 0, stream>>>(WO, Wo);
  k_qkv<<<dim3(TOKENS / 32, NQKV / 256), 256, 0, stream>>>(Xf, Wqkv, bQ, bK, bV,
                                                           Qb, Kb, Vt);
  k_attn<<<dim3(BATCH * NHEADS * (SEQ / 64)), 128, 0, stream>>>(Qb, Kb, Vt, Zb);
  k_oproj<<<dim3(TOKENS / 32, D_MODEL / 256), 256, 0, stream>>>(Zb, Wo, bO, out);
}
